// LMDGNN_8572754723471
// MI455X (gfx1250) — matrix operand order repaired and checked by reading, not compiled, not run on hardware
//
#include <hip/hip_runtime.h>
#include <hip/hip_bf16.h>

// ---------------------------------------------------------------------------
// CDNA5 (gfx1250) implementation.
//  - GEMMs via v_wmma_f32_16x16x32_f16
//  - Preferred path: inputs pre-converted to f16; tiles staged to LDS by the
//    Tensor Data Mover (tensor_load_to_lds, TENSORcnt sync), double-buffered.
//  - Fallback path (small workspace): f32 inputs, register-pipelined staging
//    with on-the-fly f16 conversion.
//  - Serial scan distributed across 16 WGPs, U matrices LDS-resident.
// ---------------------------------------------------------------------------

typedef __attribute__((ext_vector_type(16))) _Float16 v16h;
typedef __attribute__((ext_vector_type(8)))  float    v8f;
typedef __attribute__((ext_vector_type(4)))  unsigned int v4u;
typedef __attribute__((ext_vector_type(8)))  int      v8i;
typedef __attribute__((ext_vector_type(4)))  int      v4i;

union FragH { v16h v; uint4 q[2]; };

#define ACT_NONE 0
#define ACT_RELU 1
#define ACT_SIG  2

#if defined(__has_builtin)
# if __has_builtin(__builtin_amdgcn_tensor_load_to_lds)
#  define HAVE_TDM 1
# endif
# if __has_builtin(__builtin_amdgcn_s_wait_tensorcnt)
#  define HAVE_TWAIT 1
# endif
#endif

// ---------------------------------------------------------------------------
// TDM: DMA one 64-row x 64-half tile (2D) from global f16 into LDS with a
// 16-byte pad after each 128-byte row (matches _Float16 [64][72] layout).
// remK / remRows are the remaining tensor extents from the tile start;
// the TDM zero-fills out-of-bounds reads, so tails need no masking.
// ---------------------------------------------------------------------------
__device__ __forceinline__ void tdm_tile(const _Float16* base, long elemOff,
                                         int remK, int remRows,
                                         unsigned ldsOff, int ldElems)
{
#if defined(__HIP_DEVICE_COMPILE__) && defined(HAVE_TDM)
  unsigned long long ga =
      (unsigned long long)(uintptr_t)(const void*)(base + elemOff);
  v4u g0;
  g0[0] = 1u;                                    // count=1, user descriptor
  g0[1] = ldsOff;                                // lds_addr (bytes)
  g0[2] = (unsigned)(ga & 0xFFFFFFFFull);        // global_addr[31:0]
  g0[3] = (unsigned)((ga >> 32) & 0x01FFFFFFull) // global_addr[56:32]
        | (2u << 30);                            // type = 2 ("image")
  unsigned d0 = (unsigned)remK;
  unsigned d1 = (unsigned)remRows;
  v8i g1;
  g1[0] = (int)((1u << 16)        // data_size = 1 (2 bytes)
              | (1u << 20)        // pad_enable
              | (4u << 22)        // pad_interval: 32 DWORDs (=128B row)
              | (3u << 25));      // pad_amount: 4 DWORDs (=16B)
  g1[1] = (int)((d0 & 0xFFFFu) << 16);                       // tensor_dim0 lo
  g1[2] = (int)(((d0 >> 16) & 0xFFFFu) | ((d1 & 0xFFFFu) << 16));
  g1[3] = (int)(((d1 >> 16) & 0xFFFFu) | (64u << 16));       // tile_dim0 = 64
  g1[4] = (int)64;                                           // tile_dim1 = 64
  g1[5] = (int)(unsigned)ldElems;                            // dim0_stride lo
  g1[6] = 0;
  g1[7] = 0;
  v4i z4 = {0, 0, 0, 0};
#if defined(__clang_major__) && (__clang_major__ >= 23)
  v8i z8 = {0, 0, 0, 0, 0, 0, 0, 0};
  __builtin_amdgcn_tensor_load_to_lds(g0, g1, z4, z4, z8, 0);
#else
  __builtin_amdgcn_tensor_load_to_lds(g0, g1, z4, z4, 0);
#endif
#else
  (void)base; (void)elemOff; (void)remK; (void)remRows; (void)ldsOff; (void)ldElems;
#endif
}

__device__ __forceinline__ void tdm_wait0()
{
#if defined(__HIP_DEVICE_COMPILE__) && defined(HAVE_TWAIT)
  __builtin_amdgcn_s_wait_tensorcnt(0);
#endif
}

// ---------------------------------------------------------------------------
// Preferred GEMM: A,B are f16 (K-major). Tiles staged by TDM, double-buffered,
// one barrier per K-chunk. Output f32 (Cf) or f16 (Ch).
// C[m,n] = act( sum_k A[m*lda+k]*B[n*ldb+k] + bias[n] (+bias2[n]) )
// ---------------------------------------------------------------------------
__global__ __launch_bounds__(256) void gemm_f16_wmma_tdm(
    const _Float16* __restrict__ A, int lda,
    const _Float16* __restrict__ B, int ldb,
    const float* __restrict__ bias, const float* __restrict__ bias2,
    float* __restrict__ Cf, _Float16* __restrict__ Ch, int ldc,
    int M, int N, int K, int act)
{
  __shared__ alignas(16) _Float16 lsA[2][64][72];
  __shared__ alignas(16) _Float16 lsB[2][64][72];

  const int tid  = threadIdx.x;
  const int lane = tid & 31;
  const int wave = tid >> 5;
  const int rowBase = blockIdx.y * 64;
  const int colBase = blockIdx.x * 64;
  const int wm = (wave & 3) * 16;
  const int n0 = (wave >> 2) * 32;
  const int hi = lane >> 4;
  const int lm = lane & 15;

  const int nChunks = (K + 63) >> 6;

  const unsigned offA0 = (unsigned)(uintptr_t)(void*)&lsA[0][0][0];
  const unsigned offA1 = (unsigned)(uintptr_t)(void*)&lsA[1][0][0];
  const unsigned offB0 = (unsigned)(uintptr_t)(void*)&lsB[0][0][0];
  const unsigned offB1 = (unsigned)(uintptr_t)(void*)&lsB[1][0][0];

  if (wave == 0) {
    tdm_tile(A, (long)rowBase * lda, K, M - rowBase, offA0, lda);
    tdm_tile(B, (long)colBase * ldb, K, N - colBase, offB0, ldb);
  }

  v8f acc0 = {};
  v8f acc1 = {};
  int buf = 0;

  for (int i = 0; i < nChunks; ++i) {
    if (wave == 0) tdm_wait0();     // current buffer's DMA complete
    __syncthreads();                // publish LDS to all waves

    if ((wave == 0) && (i + 1 < nChunks)) {   // DMA next chunk into other buf
      int kc = (i + 1) * 64;
      tdm_tile(A, (long)rowBase * lda + kc, K - kc, M - rowBase,
               buf ? offA0 : offA1, lda);
      tdm_tile(B, (long)colBase * ldb + kc, K - kc, N - colBase,
               buf ? offB0 : offB1, ldb);
    }

    #pragma unroll
    for (int kk = 0; kk < 64; kk += 32) {
      FragH fa, fb0, fb1;
      fa.q[0]  = *(const uint4*)&lsA[buf][wm + lm][kk + hi * 16];
      fa.q[1]  = *(const uint4*)&lsA[buf][wm + lm][kk + hi * 16 + 8];
      fb0.q[0] = *(const uint4*)&lsB[buf][n0 + lm][kk + hi * 16];
      fb0.q[1] = *(const uint4*)&lsB[buf][n0 + lm][kk + hi * 16 + 8];
      fb1.q[0] = *(const uint4*)&lsB[buf][n0 + 16 + lm][kk + hi * 16];
      fb1.q[1] = *(const uint4*)&lsB[buf][n0 + 16 + lm][kk + hi * 16 + 8];
      acc0 = __builtin_amdgcn_wmma_f32_16x16x32_f16(false, fa.v, false, fb0.v,
                                                    (short)0, acc0, false, false);
      acc1 = __builtin_amdgcn_wmma_f32_16x16x32_f16(false, fa.v, false, fb1.v,
                                                    (short)0, acc1, false, false);
    }
    buf ^= 1;
  }

  #pragma unroll
  for (int tn = 0; tn < 2; ++tn) {
    v8f acc = tn ? acc1 : acc0;
    int n = colBase + n0 + tn * 16 + lm;
    if (n < N) {
      float bv = bias[n] + (bias2 ? bias2[n] : 0.0f);
      #pragma unroll
      for (int r = 0; r < 8; ++r) {
        int m = rowBase + wm + hi * 8 + r;
        float v = acc[r] + bv;
        if (act == ACT_RELU)      v = fmaxf(v, 0.0f);
        else if (act == ACT_SIG)  v = 1.0f / (1.0f + __expf(-v));
        if (Ch) Ch[(long)m * (long)ldc + n] = (_Float16)v;
        else    Cf[(long)m * (long)ldc + n] = v;
      }
    }
  }
}

// ---------------------------------------------------------------------------
// Fallback GEMM (small workspace): f32 inputs, register-pipelined staging with
// on-the-fly f16 conversion (same WMMA core). Branch-free bounds handling.
// ---------------------------------------------------------------------------
__device__ __forceinline__ void stage_load(
    const float* __restrict__ A, long arow, bool a4,
    const float* __restrict__ B, long brow, bool bok, bool b4,
    int kc, int sk, int K,
    float (&ra)[16], float (&rb)[16])
{
  const bool full = (kc + 64) <= K;
  if (a4 && full) {
    const float4* p = (const float4*)(A + arow + kc + sk);
    #pragma unroll
    for (int q = 0; q < 4; ++q) {
      float4 f = p[q];
      ra[q * 4 + 0] = f.x; ra[q * 4 + 1] = f.y;
      ra[q * 4 + 2] = f.z; ra[q * 4 + 3] = f.w;
    }
  } else if (full) {
    #pragma unroll
    for (int e = 0; e < 16; ++e) ra[e] = A[arow + kc + sk + e];
  } else {
    #pragma unroll
    for (int e = 0; e < 16; ++e) {
      int k   = kc + sk + e;
      int kcl = (k < K) ? k : (K - 1);
      float v = A[arow + kcl];
      ra[e] = (k < K) ? v : 0.0f;
    }
  }
  #pragma unroll
  for (int e = 0; e < 16; ++e) rb[e] = 0.0f;
  if (bok) {
    if (b4 && full) {
      const float4* p = (const float4*)(B + brow + kc + sk);
      #pragma unroll
      for (int q = 0; q < 4; ++q) {
        float4 f = p[q];
        rb[q * 4 + 0] = f.x; rb[q * 4 + 1] = f.y;
        rb[q * 4 + 2] = f.z; rb[q * 4 + 3] = f.w;
      }
    } else if (full) {
      #pragma unroll
      for (int e = 0; e < 16; ++e) rb[e] = B[brow + kc + sk + e];
    } else {
      #pragma unroll
      for (int e = 0; e < 16; ++e) {
        int k   = kc + sk + e;
        int kcl = (k < K) ? k : (K - 1);
        float v = B[brow + kcl];
        rb[e] = (bok && k < K) ? v : 0.0f;
      }
    }
  }
}

__global__ __launch_bounds__(256) void gemm_tn_wmma(
    const float* __restrict__ A, int lda,
    const float* __restrict__ B, int ldb,
    const float* __restrict__ bias, const float* __restrict__ bias2,
    float* __restrict__ C, int ldc,
    int M, int N, int K, int act)
{
  __shared__ alignas(16) _Float16 lsA[2][64][72];
  __shared__ alignas(16) _Float16 lsB[2][64][72];

  const int tid  = threadIdx.x;
  const int lane = tid & 31;
  const int wave = tid >> 5;
  const int rowBase = blockIdx.y * 64;
  const int colBase = blockIdx.x * 64;
  const int wm = (wave & 3) * 16;
  const int n0 = (wave >> 2) * 32;
  const int hi = lane >> 4;
  const int lm = lane & 15;

  const bool a4 = ((lda & 3) == 0) && ((((uintptr_t)A) & 15) == 0);
  const bool b4 = ((ldb & 3) == 0) && ((((uintptr_t)B) & 15) == 0);

  const int sr = tid >> 2;
  const int sk = (tid & 3) * 16;
  const long arow = (long)(rowBase + sr) * (long)lda;
  const int  bRow = colBase + sr;
  const bool bok  = bRow < N;
  const long brow = (long)(bok ? bRow : (N - 1)) * (long)ldb;

  const int nChunks = (K + 63) >> 6;

  float ra[16], rb[16];
  stage_load(A, arow, a4, B, brow, bok, b4, 0, sk, K, ra, rb);

  v8f acc0 = {};
  v8f acc1 = {};
  int buf = 0;

  for (int i = 0; i < nChunks; ++i) {
    {
      alignas(16) _Float16 ta[16], tb[16];
      #pragma unroll
      for (int e = 0; e < 16; ++e) {
        ta[e] = (_Float16)ra[e];
        tb[e] = (_Float16)rb[e];
      }
      *(uint4*)&lsA[buf][sr][sk]     = *(const uint4*)&ta[0];
      *(uint4*)&lsA[buf][sr][sk + 8] = *(const uint4*)&ta[8];
      *(uint4*)&lsB[buf][sr][sk]     = *(const uint4*)&tb[0];
      *(uint4*)&lsB[buf][sr][sk + 8] = *(const uint4*)&tb[8];
    }
    __syncthreads();

    if (i + 1 < nChunks)
      stage_load(A, arow, a4, B, brow, bok, b4, (i + 1) * 64, sk, K, ra, rb);
    if ((i + 2) * 64 + sk < K) {
      __builtin_prefetch(A + arow + (i + 2) * 64 + sk, 0, 1);
      __builtin_prefetch(B + brow + (i + 2) * 64 + sk, 0, 1);
    }

    #pragma unroll
    for (int kk = 0; kk < 64; kk += 32) {
      FragH fa, fb0, fb1;
      fa.q[0]  = *(const uint4*)&lsA[buf][wm + lm][kk + hi * 16];
      fa.q[1]  = *(const uint4*)&lsA[buf][wm + lm][kk + hi * 16 + 8];
      fb0.q[0] = *(const uint4*)&lsB[buf][n0 + lm][kk + hi * 16];
      fb0.q[1] = *(const uint4*)&lsB[buf][n0 + lm][kk + hi * 16 + 8];
      fb1.q[0] = *(const uint4*)&lsB[buf][n0 + 16 + lm][kk + hi * 16];
      fb1.q[1] = *(const uint4*)&lsB[buf][n0 + 16 + lm][kk + hi * 16 + 8];
      acc0 = __builtin_amdgcn_wmma_f32_16x16x32_f16(false, fa.v, false, fb0.v,
                                                    (short)0, acc0, false, false);
      acc1 = __builtin_amdgcn_wmma_f32_16x16x32_f16(false, fa.v, false, fb1.v,
                                                    (short)0, acc1, false, false);
    }
    buf ^= 1;
  }

  #pragma unroll
  for (int tn = 0; tn < 2; ++tn) {
    v8f acc = tn ? acc1 : acc0;
    int n = colBase + n0 + tn * 16 + lm;
    if (n < N) {
      float bv = bias[n] + (bias2 ? bias2[n] : 0.0f);
      #pragma unroll
      for (int r = 0; r < 8; ++r) {
        int m = rowBase + wm + hi * 8 + r;
        float v = acc[r] + bv;
        if (act == ACT_RELU)      v = fmaxf(v, 0.0f);
        else if (act == ACT_SIG)  v = 1.0f / (1.0f + __expf(-v));
        C[(long)m * (long)ldc + n] = v;
      }
    }
  }
}

// f32 -> f16 elementwise (handles the odd lda of x)
__global__ __launch_bounds__(256) void cvt_f32_f16(
    const float* __restrict__ src, long ldS,
    _Float16* __restrict__ dst, long ldD, int cols)
{
  int c = blockIdx.x * 256 + threadIdx.x;
  int r = blockIdx.y;
  if (c < cols) dst[(long)r * ldD + c] = (_Float16)src[(long)r * ldS + c];
}

// ---------------------------------------------------------------------------
// Distributed recurrence: 16 workgroups (one per WGP); WG w owns j in
// [16w,16w+16), keeps its 4x16x256 U slice in LDS for the whole scan, and
// exchanges 16 h-values per step via agent-scope atomics + arrive counter.
// ---------------------------------------------------------------------------
#define RGROUPS 16
#define TSTEPS 1024
#define LAGK 50

__global__ __launch_bounds__(256) void recur_kernel(
    const float* __restrict__ Ud, const float* __restrict__ Ui,
    const float* __restrict__ Uo, const float* __restrict__ Uc,
    const float* __restrict__ xw, float* __restrict__ hs,
    float* __restrict__ hb, int* __restrict__ cnt)
{
  __shared__ alignas(16) float Us[4][16][256];
  __shared__ alignas(16) float sh[256];
  __shared__ float part[4][16][4];
  __shared__ float sgate[4][16];
  __shared__ float shist[LAGK][16];

  const int tid   = threadIdx.x;
  const int jbase = blockIdx.x * 16;

  #pragma unroll
  for (int g = 0; g < 4; ++g) {
    const float* Ug = (g == 0) ? Ud : (g == 1) ? Ui : (g == 2) ? Uo : Uc;
    for (int idx = tid; idx < 16 * 256; idx += 256) {
      int j16 = idx >> 8, k = idx & 255;
      Us[g][j16][k] = Ug[(size_t)(jbase + j16) * 256 + k];
    }
  }
  for (int idx = tid; idx < LAGK * 16; idx += 256) (&shist[0][0])[idx] = 0.0f;
  __syncthreads();

  const int mg = tid >> 6;
  const int mj = (tid >> 2) & 15;
  const int mk = tid & 3;
  const float4* Urow4 = (const float4*)&Us[mg][mj][mk * 64];

  for (int t = 0; t < TSTEPS; ++t) {
    sh[tid] = __hip_atomic_load(&hb[(t & 1) * 256 + tid],
                                __ATOMIC_RELAXED, __HIP_MEMORY_SCOPE_AGENT);
    __syncthreads();

    float acc = 0.0f;
    #pragma unroll 4
    for (int q = 0; q < 16; ++q) {
      float4 u  = Urow4[q];
      float4 hv = *(const float4*)&sh[mk * 64 + q * 4];
      acc = fmaf(u.x, hv.x, acc);
      acc = fmaf(u.y, hv.y, acc);
      acc = fmaf(u.z, hv.z, acc);
      acc = fmaf(u.w, hv.w, acc);
    }
    part[mg][mj][mk] = acc;
    __syncthreads();

    if (tid < 64) {
      int g = tid >> 4, j16 = tid & 15;
      float s = part[g][j16][0] + part[g][j16][1] + part[g][j16][2] + part[g][j16][3];
      s += xw[((size_t)g * TSTEPS + t) * 256 + jbase + j16];
      float av;
      if (g == 3) av = tanhf(s);
      else { av = 1.0f / (1.0f + __expf(-s)); if (g == 0) av *= 0.5f; }
      sgate[g][j16] = av;
    }
    __syncthreads();

    if (tid < 16) {
      int j16 = tid;
      float dv = sgate[0][j16], iv = sgate[1][j16], ov = sgate[2][j16], cv = sgate[3][j16];
      int pos = t % LAGK;
      shist[pos][j16] = iv * cv;
      float w = 1.0f, mem = 0.0f;
      int idx = pos;
      #pragma unroll
      for (int k = 0; k < LAGK; ++k) {
        mem = fmaf(w, shist[idx][j16], mem);
        w *= (dv + (float)k) / (float)(k + 1);
        idx = (idx == 0) ? (LAGK - 1) : (idx - 1);
      }
      float hn = ov * tanhf(mem);
      hs[(size_t)t * 256 + jbase + j16] = hn;
      __hip_atomic_store(&hb[((t + 1) & 1) * 256 + jbase + j16], hn,
                         __ATOMIC_RELAXED, __HIP_MEMORY_SCOPE_AGENT);
    }
    __threadfence();
    __syncthreads();
    if (tid == 0) {
      __hip_atomic_fetch_add(cnt, 1, __ATOMIC_RELEASE, __HIP_MEMORY_SCOPE_AGENT);
      const int target = RGROUPS * (t + 1);
      while (__hip_atomic_load(cnt, __ATOMIC_ACQUIRE, __HIP_MEMORY_SCOPE_AGENT) < target) {
        __builtin_amdgcn_s_sleep(2);
      }
    }
    __syncthreads();
  }
}

__global__ void init_sync_kernel(float* hb, int* cnt)
{
  int i = threadIdx.x;
  if (i < 512) hb[i] = 0.0f;
  if (i == 0)  *cnt = 0;
}

// ---------------------------------------------------------------------------
extern "C" void kernel_launch(void* const* d_in, const int* in_sizes, int n_in,
                              void* d_out, int out_size, void* d_ws, size_t ws_size,
                              hipStream_t stream)
{
  (void)in_sizes; (void)n_in; (void)out_size;

  const float* x      = (const float*)d_in[0];
  const float* enc_w1 = (const float*)d_in[1];
  const float* enc_b1 = (const float*)d_in[2];
  const float* enc_w2 = (const float*)d_in[3];
  const float* enc_b2 = (const float*)d_in[4];
  const float* u_w[4], *u_b[4], *w_w[4], *w_b[4];
  for (int g = 0; g < 4; ++g) {
    u_w[g] = (const float*)d_in[5 + 4 * g + 0];
    u_b[g] = (const float*)d_in[5 + 4 * g + 1];
    w_w[g] = (const float*)d_in[5 + 4 * g + 2];
    w_b[g] = (const float*)d_in[5 + 4 * g + 3];
  }
  const float* dec_w1 = (const float*)d_in[21];
  const float* dec_b1 = (const float*)d_in[22];
  const float* dec_w2 = (const float*)d_in[23];
  const float* dec_b2 = (const float*)d_in[24];
  float* out = (float*)d_out;

  const int Bb = 1024, Nn = 50000, Hh = 512, Ee = 256;
  dim3 blk(256);

  // bump allocator over d_ws
  size_t off = 0;
  auto alloc = [&](size_t bytes) -> void* {
    void* p = (char*)d_ws + off;
    off += (bytes + 63) & ~(size_t)63;
    return p;
  };

  // sync block first (shared by both paths)
  float* hb  = (float*)alloc(512 * sizeof(float) + 64);
  int*   cnt = (int*)((char*)hb + 512 * sizeof(float));
  // common f32 buffers
  float* xw  = (float*)alloc((size_t)4 * Bb * Ee * 4);
  float* hsb = (float*)alloc((size_t)Bb * Ee * 4);

  // TDM path working set (f16 copies + f16 intermediates)
  size_t needTDM = off
      + ((size_t)Bb * Nn * 2 + 64)    // xh
      + ((size_t)Hh * Nn * 2 + 64)    // w1h
      + ((size_t)Ee * Hh * 2 + 64)    // w2h
      + 4 * ((size_t)Ee * Ee * 2 + 64)// wgh
      + ((size_t)Hh * Ee * 2 + 64)    // dw1h
      + ((size_t)Nn * Hh * 2 + 64)    // dw2h
      + ((size_t)Bb * Hh * 2 + 64)    // h1h
      + ((size_t)Bb * Ee * 2 + 64)    // ebufh
      + ((size_t)Bb * Ee * 2 + 64)    // hsh
      + ((size_t)Bb * Hh * 2 + 64);   // d1h

  init_sync_kernel<<<1, 512, 0, stream>>>(hb, cnt);

  if (ws_size >= needTDM) {
    // ---------------- TDM / f16 path ----------------
    _Float16* xh    = (_Float16*)alloc((size_t)Bb * Nn * 2);
    _Float16* w1h   = (_Float16*)alloc((size_t)Hh * Nn * 2);
    _Float16* w2h   = (_Float16*)alloc((size_t)Ee * Hh * 2);
    _Float16* wgh[4];
    for (int g = 0; g < 4; ++g) wgh[g] = (_Float16*)alloc((size_t)Ee * Ee * 2);
    _Float16* dw1h  = (_Float16*)alloc((size_t)Hh * Ee * 2);
    _Float16* dw2h  = (_Float16*)alloc((size_t)Nn * Hh * 2);
    _Float16* h1h   = (_Float16*)alloc((size_t)Bb * Hh * 2);
    _Float16* ebufh = (_Float16*)alloc((size_t)Bb * Ee * 2);
    _Float16* hsh   = (_Float16*)alloc((size_t)Bb * Ee * 2);
    _Float16* d1h   = (_Float16*)alloc((size_t)Bb * Hh * 2);

    // one-time f16 conversions
    cvt_f32_f16<<<dim3((Nn + 255) / 256, Bb), blk, 0, stream>>>(x, Nn + 1, xh, Nn, Nn);
    cvt_f32_f16<<<dim3((Nn + 255) / 256, Hh), blk, 0, stream>>>(enc_w1, Nn, w1h, Nn, Nn);
    cvt_f32_f16<<<dim3((Hh + 255) / 256, Ee), blk, 0, stream>>>(enc_w2, Hh, w2h, Hh, Hh);
    for (int g = 0; g < 4; ++g)
      cvt_f32_f16<<<dim3(1, Ee), blk, 0, stream>>>(w_w[g], Ee, wgh[g], Ee, Ee);
    cvt_f32_f16<<<dim3(1, Hh), blk, 0, stream>>>(dec_w1, Ee, dw1h, Ee, Ee);
    cvt_f32_f16<<<dim3(2, Nn), blk, 0, stream>>>(dec_w2, Hh, dw2h, Hh, Hh);

    gemm_f16_wmma_tdm<<<dim3(Hh / 64, Bb / 64), blk, 0, stream>>>(
        xh, Nn, w1h, Nn, enc_b1, nullptr, nullptr, h1h, Hh, Bb, Hh, Nn, ACT_RELU);
    gemm_f16_wmma_tdm<<<dim3(Ee / 64, Bb / 64), blk, 0, stream>>>(
        h1h, Hh, w2h, Hh, enc_b2, nullptr, nullptr, ebufh, Ee, Bb, Ee, Hh, ACT_RELU);
    for (int g = 0; g < 4; ++g)
      gemm_f16_wmma_tdm<<<dim3(Ee / 64, Bb / 64), blk, 0, stream>>>(
          ebufh, Ee, wgh[g], Ee, w_b[g], u_b[g],
          xw + (size_t)g * Bb * Ee, nullptr, Ee, Bb, Ee, Ee, ACT_NONE);

    recur_kernel<<<RGROUPS, 256, 0, stream>>>(
        u_w[0], u_w[1], u_w[2], u_w[3], xw, hsb, hb, cnt);

    cvt_f32_f16<<<dim3(1, Bb), blk, 0, stream>>>(hsb, Ee, hsh, Ee, Ee);
    gemm_f16_wmma_tdm<<<dim3(Hh / 64, Bb / 64), blk, 0, stream>>>(
        hsh, Ee, dw1h, Ee, dec_b1, nullptr, nullptr, d1h, Hh, Bb, Hh, Ee, ACT_RELU);
    gemm_f16_wmma_tdm<<<dim3((Nn + 63) / 64, Bb / 64), blk, 0, stream>>>(
        d1h, Hh, dw2h, Hh, dec_b2, nullptr, out, nullptr, Nn, Bb, Nn, Hh, ACT_SIG);
  } else {
    // ---------------- fallback f32 path ----------------
    float* h1   = (float*)alloc((size_t)Bb * Hh * 4);
    float* ebuf = (float*)alloc((size_t)Bb * Ee * 4);
    float* d1   = (float*)alloc((size_t)Bb * Hh * 4);

    gemm_tn_wmma<<<dim3(Hh / 64, Bb / 64), blk, 0, stream>>>(
        x, Nn + 1, enc_w1, Nn, enc_b1, nullptr, h1, Hh, Bb, Hh, Nn, ACT_RELU);
    gemm_tn_wmma<<<dim3(Ee / 64, Bb / 64), blk, 0, stream>>>(
        h1, Hh, enc_w2, Hh, enc_b2, nullptr, ebuf, Ee, Bb, Ee, Hh, ACT_RELU);
    for (int g = 0; g < 4; ++g)
      gemm_tn_wmma<<<dim3(Ee / 64, Bb / 64), blk, 0, stream>>>(
          ebuf, Ee, w_w[g], Ee, w_b[g], u_b[g],
          xw + (size_t)g * Bb * Ee, Ee, Bb, Ee, Ee, ACT_NONE);
    recur_kernel<<<RGROUPS, 256, 0, stream>>>(
        u_w[0], u_w[1], u_w[2], u_w[3], xw, hsb, hb, cnt);
    gemm_tn_wmma<<<dim3(Hh / 64, Bb / 64), blk, 0, stream>>>(
        hsb, Ee, dec_w1, Ee, dec_b1, nullptr, d1, Hh, Bb, Hh, Ee, ACT_RELU);
    gemm_tn_wmma<<<dim3((Nn + 63) / 64, Bb / 64), blk, 0, stream>>>(
        d1, Hh, dec_w2, Hh, dec_b2, nullptr, out, Nn, Bb, Nn, Hh, ACT_SIG);
  }
}
